// GPT_27616639714135
// MI455X (gfx1250) — matrix operand order repaired and checked by reading, not compiled, not run on hardware
//
#include <hip/hip_runtime.h>
#include <hip/hip_bf16.h>

// ---------------------------------------------------------------------------
// Transformer forward for gfx1250.
// - all dense math through v_wmma_f32_16x16x32_bf16
// - weights pre-converted/pre-transposed to bf16 once per call
// - double-buffered LDS staging via global_load_async_to_lds_b128 (ASYNCcnt)
// Workspace requirement: ~60 MB.
// ---------------------------------------------------------------------------

typedef __bf16 bf16_t;
typedef bf16_t v4bf  __attribute__((ext_vector_type(4)));
typedef bf16_t v8bf  __attribute__((ext_vector_type(8)));
typedef bf16_t v16bf __attribute__((ext_vector_type(16)));
typedef float  v8f   __attribute__((ext_vector_type(8)));

#define E_SZ   1024
#define FF_SZ  4096
#define HEADS  16
#define HDIM   64
#define BATCH  2
#define SEQ    1024
#define NXTOK  1023

__device__ __forceinline__ bf16_t f2bf(float f) { return (bf16_t)f; }

__device__ __forceinline__ v8f zero8() {
  v8f z = {0.f, 0.f, 0.f, 0.f, 0.f, 0.f, 0.f, 0.f};
  return z;
}

// --- CDNA5 async global->LDS copy (16B per lane), tracked by ASYNCcnt -------
__device__ __forceinline__ unsigned lds_off(const void* p) {
  return (unsigned)(uintptr_t)p;  // LDS aperture: addr[31:0] is the LDS offset
}
__device__ __forceinline__ void async_b128(unsigned lds, const void* g) {
  unsigned long long ga = (unsigned long long)g;
  asm volatile("global_load_async_to_lds_b128 %0, %1, off"
               :: "v"(lds), "v"(ga) : "memory");
}
__device__ __forceinline__ void wait_async() {
  asm volatile("s_wait_asynccnt 0x0" ::: "memory");
}

// A fragment (16x32 bf16): lanes 0-15 hold M=lane; K groups split by lane half.
__device__ __forceinline__ v16bf load_a_frag(const bf16_t* base, int lda, int lane) {
  int row = lane & 15;
  int kh  = (lane >> 4) & 1;
  const bf16_t* p = base + row * lda;
  v8bf lo = *(const v8bf*)(p + kh * 8);
  v8bf hi = *(const v8bf*)(p + 16 + kh * 8);
  return __builtin_shufflevector(lo, hi, 0,1,2,3,4,5,6,7,8,9,10,11,12,13,14,15);
}

// B fragment (32x16 bf16) from transposed LDS tile BT[n][k].
__device__ __forceinline__ v16bf load_bt_frag(const bf16_t* base, int ldb, int lane) {
  int col = lane & 15;
  int kh  = (lane >> 4) & 1;
  const bf16_t* p = base + col * ldb;
  v8bf lo = *(const v8bf*)(p + kh * 8);
  v8bf hi = *(const v8bf*)(p + 16 + kh * 8);
  return __builtin_shufflevector(lo, hi, 0,1,2,3,4,5,6,7,8,9,10,11,12,13,14,15);
}

__device__ __forceinline__ v8f wmma_bf16(v16bf a, v16bf b, v8f c) {
  return __builtin_amdgcn_wmma_f32_16x16x32_bf16(false, a, false, b, (short)0, c,
                                                 false, false);
}

// ---------------------------------------------------------------------------
// Weight convert+transpose: W fp32 [K][N] -> WT bf16 [N][K].  64x64 LDS tiles.
// ---------------------------------------------------------------------------
__global__ __launch_bounds__(256)
void tconv_kernel(const float* __restrict__ W, bf16_t* __restrict__ WT,
                  int K, int N) {
  __shared__ alignas(16) bf16_t tile[64 * 64];
  const int n0 = blockIdx.x * 64, k0 = blockIdx.y * 64;
  const int tid = threadIdx.x;
#pragma unroll
  for (int it = 0; it < 16; ++it) {
    int idx = tid + it * 256;
    int r = idx >> 6, c = idx & 63;
    tile[r * 64 + c] = f2bf(W[(size_t)(k0 + r) * N + n0 + c]);
  }
  __syncthreads();
#pragma unroll
  for (int it = 0; it < 4; ++it) {
    int quad = tid + it * 256;
    int n = quad & 63, kq = quad >> 6;
    v4bf o;
    o[0] = tile[(kq * 4 + 0) * 64 + n];
    o[1] = tile[(kq * 4 + 1) * 64 + n];
    o[2] = tile[(kq * 4 + 2) * 64 + n];
    o[3] = tile[(kq * 4 + 3) * 64 + n];
    *(v4bf*)(WT + (size_t)(n0 + n) * K + k0 + kq * 4) = o;
  }
}

// ---------------------------------------------------------------------------
// GEMM: C[M,N] = act(A[M,K] @ W[K,N] + bias[N]),  W given as WT bf16 [N][K].
// OUT: 0 = f32, 1 = bf16, 2 = bf16+relu, 3 = bf16 transposed-per-head (V).
// ABF: A matrix is bf16 (async-staged) instead of fp32 (convert-staged).
// BK=64, double-buffered LDS; next tile prefetched while WMMAs execute.
// ---------------------------------------------------------------------------
#define BKG   64
#define LDA_S 72
#define LDB_S 72

template <int OUT, int ABF>
__global__ __launch_bounds__(256)
void gemm_bf16(const void* __restrict__ Ain, const bf16_t* __restrict__ Wt,
               const float* __restrict__ bias, void* __restrict__ Cout,
               int M, int N, int K) {
  __shared__ alignas(16) bf16_t As[2][128 * LDA_S];
  __shared__ alignas(16) bf16_t Bs[2][128 * LDB_S];

  const int tid  = threadIdx.x;
  const int lane = tid & 31;
  const int wave = tid >> 5;
  const int wrow = wave & 3;   // 4 groups of 32 rows
  const int wcol = wave >> 2;  // 2 groups of 64 cols
  const int m0 = blockIdx.y * 128;
  const int n0 = blockIdx.x * 128;
  const unsigned as0 = lds_off(&As[0][0]);
  const unsigned bs0 = lds_off(&Bs[0][0]);
  const unsigned ABUF = 128 * LDA_S * 2;  // bytes per buffer
  const unsigned BBUF = 128 * LDB_S * 2;

  auto stage_async = [&](int buf, int k0) {
    const unsigned bb = bs0 + (unsigned)buf * BBUF;
#pragma unroll
    for (int it = 0; it < 4; ++it) {
      int c2 = tid + it * 256;  // 1024 chunks of 16B
      int n = c2 >> 3, ch = c2 & 7;
      async_b128(bb + n * (LDB_S * 2) + ch * 16,
                 Wt + (size_t)(n0 + n) * K + k0 + ch * 8);
    }
    if (ABF) {
      const unsigned ab = as0 + (unsigned)buf * ABUF;
      const bf16_t* Ab = (const bf16_t*)Ain;
#pragma unroll
      for (int it = 0; it < 4; ++it) {
        int c2 = tid + it * 256;
        int r = c2 >> 3, ch = c2 & 7;
        async_b128(ab + r * (LDA_S * 2) + ch * 16,
                   Ab + (size_t)(m0 + r) * K + k0 + ch * 8);
      }
    }
  };
  auto stage_convA = [&](int buf, int k0) {
    if (!ABF) {
      const float* Af = (const float*)Ain;
      bf16_t* Ad = &As[buf][0];
#pragma unroll
      for (int it = 0; it < 8; ++it) {
        int idx = tid + it * 256;  // 2048 float4 units
        int r = idx >> 4, c4 = idx & 15;
        const float4 val = *(const float4*)(Af + (size_t)(m0 + r) * K + k0 + c4 * 4);
        v4bf o;
        o[0] = f2bf(val.x); o[1] = f2bf(val.y);
        o[2] = f2bf(val.z); o[3] = f2bf(val.w);
        *(v4bf*)(&Ad[r * LDA_S + c4 * 4]) = o;
      }
    }
  };

  v8f acc[2][4];
#pragma unroll
  for (int i = 0; i < 2; ++i)
#pragma unroll
    for (int j = 0; j < 4; ++j) acc[i][j] = zero8();

  // prologue: fill buffer 0
  stage_async(0, 0);
  stage_convA(0, 0);
  wait_async();
  __syncthreads();

  for (int k0 = 0; k0 < K; k0 += BKG) {
    const int cur = (k0 >> 6) & 1;
    const int nxt = cur ^ 1;
    const bool pf = (k0 + BKG) < K;
    if (pf) stage_async(nxt, k0 + BKG);  // async prefetch issues first

#pragma unroll
    for (int kc = 0; kc < 2; ++kc) {
      v16bf af[2], bfr[4];
#pragma unroll
      for (int mt = 0; mt < 2; ++mt)
        af[mt] = load_a_frag(&As[cur][(wrow * 32 + mt * 16) * LDA_S + kc * 32],
                             LDA_S, lane);
#pragma unroll
      for (int nt = 0; nt < 4; ++nt)
        bfr[nt] = load_bt_frag(&Bs[cur][(wcol * 64 + nt * 16) * LDB_S + kc * 32],
                               LDB_S, lane);
#pragma unroll
      for (int mt = 0; mt < 2; ++mt)
#pragma unroll
        for (int nt = 0; nt < 4; ++nt)
          acc[mt][nt] = wmma_bf16(af[mt], bfr[nt], acc[mt][nt]);
    }

    if (pf) stage_convA(nxt, k0 + BKG);  // cvt VALU co-executes with WMMAs
    wait_async();
    __syncthreads();
  }

  const int mrow = ((lane >> 4) & 1) * 8;
#pragma unroll
  for (int mt = 0; mt < 2; ++mt) {
#pragma unroll
    for (int nt = 0; nt < 4; ++nt) {
      int col = n0 + wcol * 64 + nt * 16 + (lane & 15);
      float bb = bias[col];
      int rbase = m0 + wrow * 32 + mt * 16 + mrow;
#pragma unroll
      for (int r = 0; r < 8; ++r) {
        float vv = acc[mt][nt][r] + bb;
        if (OUT == 2) vv = fmaxf(vv, 0.f);
        int row = rbase + r;
        if (OUT == 0) {
          ((float*)Cout)[(size_t)row * N + col] = vv;
        } else if (OUT == 3) {
          int b = row >> 10, s = row & 1023;   // V output: [b][h][hd][s]
          int h = col >> 6, hd = col & 63;
          ((bf16_t*)Cout)[((((size_t)b * HEADS + h) * HDIM + hd) << 10) + s] =
              f2bf(vv);
        } else {
          ((bf16_t*)Cout)[(size_t)row * N + col] = f2bf(vv);
        }
      }
    }
  }
}

// ---------------------------------------------------------------------------
// Attention: one block = (b, h, 16 query rows); 4 waves.
// Q,K bf16 [b][s][h*64+hd]; V bf16 pre-transposed [b][h][hd][s]; O fp32.
// 128 keys staged per step, double-buffered async.  NOTE: single 2-D shared
// array indexed by buffer id keeps fragment reads on the DS path (no flat).
// ---------------------------------------------------------------------------
#define VT_LD 136  // V^T LDS row stride in elements (272B, 16B aligned)

__global__ __launch_bounds__(128)
void attn_kernel(const bf16_t* __restrict__ Q, const bf16_t* __restrict__ Kg,
                 const bf16_t* __restrict__ Vt, float* __restrict__ O,
                 int causal) {
  __shared__ alignas(16) float  sc[16 * 1024];       // raw scores fp32 (64 KB)
  __shared__ alignas(16) bf16_t ps[16 * 1024];       // probs bf16 (32 KB)
  __shared__ alignas(16) bf16_t qs[16 * 64];         // Q tile
  __shared__ alignas(16) bf16_t kbuf[2][64 * VT_LD]; // K / V^T stage buffers
  __shared__ float red[16 * 8];

  const int q0 = blockIdx.x * 16;
  const int h  = blockIdx.y;
  const int b  = blockIdx.z;
  const int tid = threadIdx.x;
  const int lane = tid & 31;
  const int wave = tid >> 5;

  const bf16_t* Qb = Q + ((size_t)b * SEQ + q0) * E_SZ + h * HDIM;
  const bf16_t* Kb = Kg + ((size_t)b * SEQ) * E_SZ + h * HDIM;
  const bf16_t* Vb = Vt + (((size_t)b * HEADS + h) * HDIM) * SEQ;
  float* Ob = O + ((size_t)b * SEQ + q0) * E_SZ + h * HDIM;
  const unsigned qsoff = lds_off(qs);
  const unsigned k0off = lds_off(&kbuf[0][0]);
  const unsigned KBUF = 64 * VT_LD * 2;  // bytes per buffer

  // K stage: [key 128][hd 64], stride 64 elems (128B)
  auto stage_k = [&](int buf, int kbase) {
    const unsigned bo = k0off + (unsigned)buf * KBUF;
#pragma unroll
    for (int j = 0; j < 8; ++j) {
      int c2 = tid + j * 128;  // 1024 chunks of 16B
      int key = c2 >> 3, ch = c2 & 7;
      async_b128(bo + key * 128 + ch * 16,
                 Kb + (size_t)(kbase + key) * E_SZ + ch * 8);
    }
  };
  // V^T stage: [hd 64][key 128], stride VT_LD elems (272B)
  auto stage_v = [&](int buf, int kbase) {
    const unsigned bo = k0off + (unsigned)buf * KBUF;
#pragma unroll
    for (int j = 0; j < 8; ++j) {
      int c2 = tid + j * 128;
      int hd = c2 >> 4, ch = c2 & 15;
      async_b128(bo + hd * (VT_LD * 2) + ch * 16,
                 Vb + (size_t)hd * SEQ + kbase + ch * 8);
    }
  };

  // stage Q tile 16x64 (128 chunks of 16B, one per thread)
  async_b128(qsoff + tid * 16, Qb + (size_t)(tid >> 3) * E_SZ + (tid & 7) * 8);
  wait_async();
  __syncthreads();
  v16bf qa0 = load_a_frag(qs + 0, 64, lane);
  v16bf qa1 = load_a_frag(qs + 32, 64, lane);

  const float inv32 = 1.0f / 32.0f;  // 1/sqrt(E) quirk
  const int mrow = ((lane >> 4) & 1) * 8;

  // ---- scores: 128 keys per stage; each wave owns 32 keys (2 tiles) ----
  stage_k(0, 0);
  wait_async();
  __syncthreads();
  for (int it = 0; it < SEQ / 128; ++it) {
    const int cur = it & 1;
    if (it < SEQ / 128 - 1) stage_k(cur ^ 1, (it + 1) * 128);
    const bf16_t* kp = &kbuf[cur][0];  // addrspace(3) GEP -> ds_load
#pragma unroll
    for (int t = 0; t < 2; ++t) {
      const bf16_t* kt = kp + (wave * 32 + t * 16) * 64;  // [key][hd] == BT
      v8f c8 = zero8();
      c8 = wmma_bf16(qa0, load_bt_frag(kt + 0, 64, lane), c8);
      c8 = wmma_bf16(qa1, load_bt_frag(kt + 32, 64, lane), c8);
      int ncol = it * 128 + wave * 32 + t * 16 + (lane & 15);
#pragma unroll
      for (int r = 0; r < 8; ++r) {
        int qrow = mrow + r;
        float s = c8[r];
        if (causal && (ncol > q0 + qrow)) s = -1e9f;  // mask BEFORE scale
        sc[qrow * 1024 + ncol] = s * inv32;
      }
    }
    wait_async();
    __syncthreads();
  }

  // ---- softmax: 8 threads per row ----
  {
    int row = tid >> 3;
    int sub = tid & 7;
    float* srow = sc + row * 1024;
    float mx = -3.4e38f;
    for (int j = sub; j < SEQ; j += 8) mx = fmaxf(mx, srow[j]);
    red[row * 8 + sub] = mx;
    __syncthreads();
    mx = red[row * 8];
#pragma unroll
    for (int i = 1; i < 8; ++i) mx = fmaxf(mx, red[row * 8 + i]);
    __syncthreads();
    float sum = 0.f;
    for (int j = sub; j < SEQ; j += 8) {
      float e = __expf(srow[j] - mx);
      srow[j] = e;
      sum += e;
    }
    red[row * 8 + sub] = sum;
    __syncthreads();
    sum = 0.f;
#pragma unroll
    for (int i = 0; i < 8; ++i) sum += red[row * 8 + i];
    float inv = 1.0f / sum;
    bf16_t* prow = ps + row * 1024;
    for (int j = sub; j < SEQ; j += 8) prow[j] = f2bf(srow[j] * inv);
  }

  // ---- O = P(16xS) @ V(S x 64): 128 keys per stage, double-buffered ----
  v8f oacc = zero8();
  stage_v(0, 0);
  wait_async();
  __syncthreads();  // also publishes ps writes
  for (int it = 0; it < SEQ / 128; ++it) {
    const int cur = it & 1;
    if (it < SEQ / 128 - 1) stage_v(cur ^ 1, (it + 1) * 128);
    const bf16_t* vt = &kbuf[cur][0];
#pragma unroll
    for (int j = 0; j < 4; ++j) {
      v16bf pa = load_a_frag(ps + it * 128 + j * 32, 1024, lane);
      v16bf vb = load_bt_frag(vt + (wave * 16) * VT_LD + j * 32, VT_LD, lane);
      oacc = wmma_bf16(pa, vb, oacc);
    }
    wait_async();
    __syncthreads();
  }
  int col = wave * 16 + (lane & 15);
#pragma unroll
  for (int r = 0; r < 8; ++r)
    Ob[(size_t)(mrow + r) * E_SZ + col] = oacc[r];
}

// ---------------------------------------------------------------------------
// Embedding: out[b,s,:] = emb[tok]*32 + pe[b,:]  (pe indexed by BATCH, quirk).
// Rows s >= n_tok get the raw cls vector.
// ---------------------------------------------------------------------------
__global__ __launch_bounds__(256)
void embed_kernel(const float* __restrict__ emb, const int* __restrict__ tok,
                  const float* __restrict__ cls, float* __restrict__ out,
                  int n_tok, int seq) {
  int s = blockIdx.x, b = blockIdx.y;
  float* o = out + ((size_t)b * seq + s) * E_SZ;
  if (s < n_tok) {
    int t = tok[b * n_tok + s];
    const float* e = emb + (size_t)t * E_SZ;
    const float kfac = 9.210340372f / 1024.0f;  // ln(10000)/E
    for (int d = threadIdx.x; d < E_SZ; d += 256) {
      int i = d >> 1;
      float arg = (float)b * __expf(-(float)(2 * i) * kfac);
      float pe = (d & 1) ? __cosf(arg) : __sinf(arg);
      o[d] = e[d] * 32.0f + pe;
    }
  } else {
    for (int d = threadIdx.x; d < E_SZ; d += 256) o[d] = cls[d];
  }
}

// ---------------------------------------------------------------------------
// LayerNorm with fused residual: out = LN(res + src) * g + b. One block/row.
// ---------------------------------------------------------------------------
__global__ __launch_bounds__(256)
void ln_kernel(const float* __restrict__ res, const float* __restrict__ src,
               const float* __restrict__ g, const float* __restrict__ bta,
               float* __restrict__ out) {
  __shared__ float s1[8], s2[8];
  size_t row = blockIdx.x;
  const float* r0 = res + row * E_SZ;
  const float* r1 = src + row * E_SZ;
  float* o = out + row * E_SZ;
  float a = 0.f, q = 0.f;
  for (int d = threadIdx.x; d < E_SZ; d += 256) {
    float v = r0[d] + r1[d];
    a += v;
    q += v * v;
  }
#pragma unroll
  for (int off = 16; off > 0; off >>= 1) {
    a += __shfl_down(a, off, 32);
    q += __shfl_down(q, off, 32);
  }
  if ((threadIdx.x & 31) == 0) { s1[threadIdx.x >> 5] = a; s2[threadIdx.x >> 5] = q; }
  __syncthreads();
  if (threadIdx.x == 0) {
    float ta = 0.f, tq = 0.f;
#pragma unroll
    for (int i = 0; i < 8; ++i) { ta += s1[i]; tq += s2[i]; }
    s1[0] = ta; s2[0] = tq;
  }
  __syncthreads();
  float mean = s1[0] * (1.0f / E_SZ);
  float var = s2[0] * (1.0f / E_SZ) - mean * mean;
  float inv = rsqrtf(var + 1e-5f);
  for (int d = threadIdx.x; d < E_SZ; d += 256) {
    float v = r0[d] + r1[d];
    o[d] = (v - mean) * inv * g[d] + bta[d];
  }
}

// ---------------------------------------------------------------------------
extern "C" void kernel_launch(void* const* d_in, const int* in_sizes, int n_in,
                              void* d_out, int out_size, void* d_ws, size_t ws_size,
                              hipStream_t stream) {
  const float* emb = (const float*)d_in[0];
  const float* cls = (const float*)d_in[1];
  const float* sWq = (const float*)d_in[2];
  const float* sbq = (const float*)d_in[3];
  const float* sWk = (const float*)d_in[4];
  const float* sbk = (const float*)d_in[5];
  const float* sWv = (const float*)d_in[6];
  const float* sbv = (const float*)d_in[7];
  const float* sWo = (const float*)d_in[8];
  const float* sbo = (const float*)d_in[9];
  const float* cWq = (const float*)d_in[10];
  const float* cbq = (const float*)d_in[11];
  const float* cWk = (const float*)d_in[12];
  const float* cbk = (const float*)d_in[13];
  const float* cWv = (const float*)d_in[14];
  const float* cbv = (const float*)d_in[15];
  const float* cWo = (const float*)d_in[16];
  const float* cbo = (const float*)d_in[17];
  const float* fW1 = (const float*)d_in[18];
  const float* fb1 = (const float*)d_in[19];
  const float* fW2 = (const float*)d_in[20];
  const float* fb2 = (const float*)d_in[21];
  const float* g1 = (const float*)d_in[22];
  const float* g2 = (const float*)d_in[23];
  const float* g3 = (const float*)d_in[24];
  const float* b1 = (const float*)d_in[25];
  const float* b2 = (const float*)d_in[26];
  const float* b3 = (const float*)d_in[27];
  const int* x_tok = (const int*)d_in[28];
  const int* y_tok = (const int*)d_in[29];

  const size_t MB = 1u << 20;
  char* base = (char*)d_ws;
  float*  y    = (float*)(base);            // 8 MB
  float*  x    = (float*)(base + 8 * MB);   // 8 MB
  float*  yl   = (float*)(base + 16 * MB);  // 8 MB
  char*   sc0  = base + 24 * MB;            // 20 MB shared scratch
  bf16_t* qb   = (bf16_t*)(sc0);            // 4 MB
  bf16_t* kb   = (bf16_t*)(sc0 + 4 * MB);   // 4 MB
  bf16_t* vtb  = (bf16_t*)(sc0 + 8 * MB);   // 4 MB
  float*  aout = (float*)(sc0 + 12 * MB);   // 8 MB
  bf16_t* hidb = (bf16_t*)(sc0);            // 16 MB (overlays q/k/vt/aout)
  float*  go   = (float*)(base + 44 * MB);  // 8 MB
  bf16_t* wt   = (bf16_t*)(base + 52 * MB); // 8 MB (largest WT)

  embed_kernel<<<dim3(SEQ, BATCH), 256, 0, stream>>>(emb, y_tok, cls, y, SEQ, SEQ);
  embed_kernel<<<dim3(SEQ, BATCH), 256, 0, stream>>>(emb, x_tok, cls, x, NXTOK, SEQ);

  const int M = BATCH * SEQ;  // 2048
  auto conv = [&](const float* W, int K, int N) {
    tconv_kernel<<<dim3(N / 64, K / 64), 256, 0, stream>>>(W, wt, K, N);
  };
  auto gemm = [&](int outmode, int abf, const void* A, const float* bias,
                  void* C, int N, int K) {
    dim3 grid(N / 128, M / 128);
    if (abf)               gemm_bf16<0, 1><<<grid, 256, 0, stream>>>(A, wt, bias, C, M, N, K);
    else if (outmode == 0) gemm_bf16<0, 0><<<grid, 256, 0, stream>>>(A, wt, bias, C, M, N, K);
    else if (outmode == 1) gemm_bf16<1, 0><<<grid, 256, 0, stream>>>(A, wt, bias, C, M, N, K);
    else if (outmode == 2) gemm_bf16<2, 0><<<grid, 256, 0, stream>>>(A, wt, bias, C, M, N, K);
    else                   gemm_bf16<3, 0><<<grid, 256, 0, stream>>>(A, wt, bias, C, M, N, K);
  };
  dim3 agrid(SEQ / 16, HEADS, BATCH);
  dim3 lgrid(M);

  for (int l = 0; l < 4; ++l) {
    const size_t we = (size_t)l * E_SZ * E_SZ;
    const size_t be = (size_t)l * E_SZ;
    // ---- self-attention on ORIGINAL y (reference quirk), causal ----
    conv(sWq + we, E_SZ, E_SZ); gemm(1, 0, y, sbq + be, qb, E_SZ, E_SZ);
    conv(sWk + we, E_SZ, E_SZ); gemm(1, 0, y, sbk + be, kb, E_SZ, E_SZ);
    conv(sWv + we, E_SZ, E_SZ); gemm(3, 0, y, sbv + be, vtb, E_SZ, E_SZ);
    attn_kernel<<<agrid, 128, 0, stream>>>(qb, kb, vtb, aout, 1);
    conv(sWo + we, E_SZ, E_SZ); gemm(0, 0, aout, sbo + be, go, E_SZ, E_SZ);
    ln_kernel<<<lgrid, 256, 0, stream>>>(y, go, g1 + be, b1 + be, yl);
    // ---- cross-attention: Q from yl, K/V from current x, no mask ----
    conv(cWq + we, E_SZ, E_SZ); gemm(1, 0, yl, cbq + be, qb, E_SZ, E_SZ);
    conv(cWk + we, E_SZ, E_SZ); gemm(1, 0, x, cbk + be, kb, E_SZ, E_SZ);
    conv(cWv + we, E_SZ, E_SZ); gemm(3, 0, x, cbv + be, vtb, E_SZ, E_SZ);
    attn_kernel<<<agrid, 128, 0, stream>>>(qb, kb, vtb, aout, 0);
    conv(cWo + we, E_SZ, E_SZ); gemm(0, 0, aout, cbo + be, go, E_SZ, E_SZ);
    ln_kernel<<<lgrid, 256, 0, stream>>>(yl, go, g2 + be, b2 + be, x);
    // ---- FFN: bf16 hidden, bf16-A second GEMM ----
    conv(fW1 + (size_t)l * E_SZ * FF_SZ, E_SZ, FF_SZ);
    gemm(2, 0, x, fb1 + (size_t)l * FF_SZ, hidb, FF_SZ, E_SZ);
    conv(fW2 + (size_t)l * FF_SZ * E_SZ, FF_SZ, E_SZ);
    gemm(0, 1, hidb, fb2 + be, go, E_SZ, FF_SZ);
    float* dest = (l == 3) ? (float*)d_out : x;
    ln_kernel<<<lgrid, 256, 0, stream>>>(x, go, g3 + be, b3 + be, dest);
  }
  (void)in_sizes; (void)n_in; (void)out_size; (void)ws_size;
}
